// NewHyperLinear_4114578670093
// MI455X (gfx1250) — matrix operand order repaired and checked by reading, not compiled, not run on hardware
//
#include <hip/hip_runtime.h>

// ---------------------------------------------------------------------------
// NewHyperLinear for MI455X (gfx1250, wave32, WMMA).
//
//   out[b,o] = sum_{i,n} (x[b,i]*z[b,n]) * hW[i*512+o, n]      (K=65536 GEMM)
//            + sum_i     x[b,i] * hb[i*512+o]                  (K=512 GEMM)
//            + sum_n     z[b,n] * hW[262144+o, n] + hb[262144+o]
//            + same with (prior_x, pW, pb)
//
// 68.7 GFLOP dominated by two K=65536 GEMMs; 269 MB of hW/pW traffic
// (11.5 us HBM floor). bf16 operands + f32 WMMA accumulate.
//
// Round-2 changes: 32x32 wave tiles (1 fragment load per WMMA instead of 3),
// double-buffered LDS (1 barrier/stage, global loads overlap WMMA),
// B staging via o-quad threads -> ds_store_b64 instead of scattered b16.
// ---------------------------------------------------------------------------

#define IN_F   512
#define OUT_F  512
#define ND     128
#define WROWS  (IN_F * OUT_F)      /* 262144 */

typedef __attribute__((ext_vector_type(16))) __bf16 v16bf;
typedef __attribute__((ext_vector_type(4)))  __bf16 v4bf;
typedef __attribute__((ext_vector_type(8)))  float  v8f;
typedef __attribute__((ext_vector_type(4)))  float  v4f;

// --- WMMA fragment-linear LDS layouts (bf16 element index) -----------------
// A (16x32, 16-bit; ISA 7.12.2): lanes 0-15 hold M with K={0..7,16..23},
// lanes 16-31 hold K={8..15,24..31}; 2 elems per VGPR.
__device__ __forceinline__ int a_idx(int b_local, int k) {
  int lane = (b_local & 15) | ((k & 8) << 1);
  int e    = (k & 7) | ((k & 16) >> 1);
  return ((b_local >> 4) * 512) + lane * 16 + e;
}
// B (32x16, 16-bit): same K order as A; lanes 0-15 hold N with K={0..7,16..23},
// lanes 16-31 hold K={8..15,24..31}.
__device__ __forceinline__ int b_idx(int k, int o_local) {
  int lane = (o_local & 15) | ((k & 8) << 1);
  int e    = (k & 7) | ((k & 16) >> 1);
  return ((o_local >> 4) * 512) + lane * 16 + e;
}

__device__ __forceinline__ v8f wmma_bf16(v16bf a, v16bf b, v8f c) {
  return __builtin_amdgcn_wmma_f32_16x16x32_bf16(false, a, false, b, (short)0,
                                                 c, false, false);
}

__global__ void zero_out_kernel(float* __restrict__ out, int n) {
  int i = blockIdx.x * 256 + threadIdx.x;
  if (i < n) out[i] = 0.0f;
}

// ---------------------------------------------------------------------------
// Main kernel: the two K=65536 GEMMs, split-K over i (grid.z = 8).
// WG tile 64(batch) x 128(out); 8 waves in 2(M) x 4(N); wave tile 32x32.
// Double-buffered LDS staging, one barrier per K=32 stage.
// ---------------------------------------------------------------------------
__global__ __launch_bounds__(256) void hyper_main_kernel(
    const float* __restrict__ x,  const float* __restrict__ px,
    const float* __restrict__ z,  const float* __restrict__ hW,
    const float* __restrict__ pW, float* __restrict__ out) {
  const int N0 = blockIdx.x * 128;
  const int M0 = blockIdx.y * 64;
  const int i0 = blockIdx.z * 64;
  const int tid = threadIdx.x, lane = tid & 31, wid = tid >> 5;
  const int wm2 = wid & 1;        // M half: rows M0 + wm2*32 ..+32
  const int wn4 = wid >> 1;       // N quarter: cols N0 + wn4*32 ..+32

  // Double buffers: A = 2 matrices x 4 M-frags x 512; B = 2 x 8 N-frags x 512
  __shared__ __align__(32) __bf16 sA[2][4096];   // 16 KB
  __shared__ __align__(32) __bf16 sB[2][8192];   // 32 KB

  v8f acc00 = {}, acc01 = {}, acc10 = {}, acc11 = {};

  // Staging roles. A: thread owns batch row bo, 8 consecutive k (contiguous
  // LDS -> merges to ds_store_b128). B: thread owns an o-quad x k-quad so the
  // 4 bf16 for (kqB..kqB+3, o) are LDS-adjacent -> ds_store_b64.
  const int bo  = tid >> 2;            // 0..63
  const int kqA = (tid & 3) * 8;       // 0,8,16,24
  const int oq  = (tid >> 3) * 4;      // 0..124 step 4
  const int kqB = (tid & 7) * 4;       // 0..28 step 4

  // Stage registers (software pipeline)
  v4f bh[4], bp[4], zva, zvb;
  float xh, xp;

  auto load_stage = [&](int s) {
    const int i  = i0 + (s >> 2);
    const int nb = (s & 3) * 32;
    const float* __restrict__ hrow = hW + (size_t)(i * OUT_F + N0 + oq) * ND + nb + kqB;
    const float* __restrict__ prow = pW + (size_t)(i * OUT_F + N0 + oq) * ND + nb + kqB;
#pragma unroll
    for (int r = 0; r < 4; ++r) {
      bh[r] = *(const v4f*)(hrow + (size_t)r * ND);
      bp[r] = *(const v4f*)(prow + (size_t)r * ND);
    }
    const float* __restrict__ zr = z + (M0 + bo) * ND + nb + kqA;
    zva = *(const v4f*)zr;
    zvb = *(const v4f*)(zr + 4);
    xh = x [(M0 + bo) * IN_F + i];
    xp = px[(M0 + bo) * IN_F + i];
  };

  auto store_stage = [&](int buf) {
#pragma unroll
    for (int j = 0; j < 4; ++j) {
      const int o = oq + j;
      v4bf qh = { (__bf16)bh[j][0], (__bf16)bh[j][1],
                  (__bf16)bh[j][2], (__bf16)bh[j][3] };
      v4bf qp = { (__bf16)bp[j][0], (__bf16)bp[j][1],
                  (__bf16)bp[j][2], (__bf16)bp[j][3] };
      *(v4bf*)&sB[buf][b_idx(kqB, o)]        = qh;
      *(v4bf*)&sB[buf][4096 + b_idx(kqB, o)] = qp;
    }
#pragma unroll
    for (int j = 0; j < 8; ++j) {
      const float zj = (j < 4) ? zva[j & 3] : zvb[j & 3];
      const int k = kqA + j;
      sA[buf][a_idx(bo, k)]        = (__bf16)(xh * zj);   // A = outer(x, z)
      sA[buf][2048 + a_idx(bo, k)] = (__bf16)(xp * zj);
    }
  };

  auto mma_stage = [&](int buf) {
    const v16bf* A = (const v16bf*)sA[buf];
    const v16bf* B = (const v16bf*)sB[buf];
    v16bf ah0 = A[(wm2 * 2 + 0) * 32 + lane];
    v16bf ah1 = A[(wm2 * 2 + 1) * 32 + lane];
    v16bf ap0 = A[128 + (wm2 * 2 + 0) * 32 + lane];
    v16bf ap1 = A[128 + (wm2 * 2 + 1) * 32 + lane];
    v16bf bh0 = B[(wn4 * 2 + 0) * 32 + lane];
    v16bf bh1 = B[(wn4 * 2 + 1) * 32 + lane];
    v16bf bp0 = B[256 + (wn4 * 2 + 0) * 32 + lane];
    v16bf bp1 = B[256 + (wn4 * 2 + 1) * 32 + lane];
    acc00 = wmma_bf16(ah0, bh0, acc00);
    acc01 = wmma_bf16(ah0, bh1, acc01);
    acc10 = wmma_bf16(ah1, bh0, acc10);
    acc11 = wmma_bf16(ah1, bh1, acc11);
    acc00 = wmma_bf16(ap0, bp0, acc00);
    acc01 = wmma_bf16(ap0, bp1, acc01);
    acc10 = wmma_bf16(ap1, bp0, acc10);
    acc11 = wmma_bf16(ap1, bp1, acc11);
  };

  // Software pipeline: stage s lives in buffer s&1. One barrier per stage.
  load_stage(0);
  store_stage(0);
  for (int s = 0; s < 256; ++s) {
    __syncthreads();                 // buf[s&1] written; buf[(s+1)&1] drained
    if (s + 1 < 256) load_stage(s + 1);    // global loads overlap WMMA below
    mma_stage(s & 1);
    if (s + 1 < 256) store_stage((s + 1) & 1);
  }

  // Writeout. C/D layout: lane<16 -> N=lane, M=r; lane>=16 -> N=lane-16, M=r+8
  const int ncol = N0 + wn4 * 32 + (lane & 15);
  const int mrow = M0 + wm2 * 32 + ((lane >> 4) << 3);
#pragma unroll
  for (int r = 0; r < 8; ++r) {
    atomicAdd(&out[(mrow + r)      * OUT_F + ncol],      acc00[r]);
    atomicAdd(&out[(mrow + r)      * OUT_F + ncol + 16], acc01[r]);
    atomicAdd(&out[(mrow + 16 + r) * OUT_F + ncol],      acc10[r]);
    atomicAdd(&out[(mrow + 16 + r) * OUT_F + ncol + 16], acc11[r]);
  }
}

// ---------------------------------------------------------------------------
// Extras kernel (<2% of FLOPs): x@HB + px@PB (K=512), z@hWtail^T + z@pWtail^T
// (K=128), plus the scalar bias tail. 64x64 tiles, 16x32 wave tiles.
// ---------------------------------------------------------------------------
__device__ __forceinline__ void mma_stage_16x32(const __bf16* sA, const __bf16* sB,
                                                int wm, int wn, int lane,
                                                v8f& acc0, v8f& acc1) {
  const v16bf* A = (const v16bf*)sA;
  const v16bf* B = (const v16bf*)sB;
  v16bf a_h  = A[wm * 32 + lane];
  v16bf a_p  = A[128 + wm * 32 + lane];
  v16bf b_h0 = B[wn * 64 + lane];
  v16bf b_h1 = B[wn * 64 + 32 + lane];
  v16bf b_p0 = B[128 + wn * 64 + lane];
  v16bf b_p1 = B[128 + wn * 64 + 32 + lane];
  acc0 = wmma_bf16(a_h, b_h0, acc0);
  acc1 = wmma_bf16(a_h, b_h1, acc1);
  acc0 = wmma_bf16(a_p, b_p0, acc0);
  acc1 = wmma_bf16(a_p, b_p1, acc1);
}

__global__ __launch_bounds__(256) void hyper_extras_kernel(
    const float* __restrict__ x,  const float* __restrict__ px,
    const float* __restrict__ z,  const float* __restrict__ hW,
    const float* __restrict__ pW, const float* __restrict__ hb,
    const float* __restrict__ pb, float* __restrict__ out) {
  const int N0 = blockIdx.x * 64;
  const int M0 = blockIdx.y * 64;
  const int tid = threadIdx.x, lane = tid & 31, wid = tid >> 5;
  const int wm = wid & 3, wn = wid >> 2;

  __shared__ __align__(32) __bf16 sA[4096];
  __shared__ __align__(32) __bf16 sB[4096];

  v8f acc0 = {}; v8f acc1 = {};

  // ---- segment 1: K = 512 over i;  B[k][o] = hb[k*512+o] (row-major) ----
  {
    const int bo = tid >> 2, kq = (tid & 3) * 8;  // A: row bo, 8 k's
    const int ko = tid >> 3, oq = (tid & 7) * 8;  // B: row ko, 8 o's
    for (int kc = 0; kc < IN_F; kc += 32) {
      __syncthreads();
#pragma unroll
      for (int j = 0; j < 8; ++j) {
        int o = oq + j;
        sB[b_idx(ko, o)]        = (__bf16)hb[(size_t)(kc + ko) * OUT_F + N0 + o];
        sB[2048 + b_idx(ko, o)] = (__bf16)pb[(size_t)(kc + ko) * OUT_F + N0 + o];
        int k = kq + j;
        sA[a_idx(bo, k)]        = (__bf16)x [(M0 + bo) * IN_F + kc + k];
        sA[2048 + a_idx(bo, k)] = (__bf16)px[(M0 + bo) * IN_F + kc + k];
      }
      __syncthreads();
      mma_stage_16x32(sA, sB, wm, wn, lane, acc0, acc1);
    }
  }
  // ---- segment 2: K = 128 over n;  B[n][o] = W[(262144+o)*128 + n] ----
  {
    const int bo = tid >> 2, kq = (tid & 3) * 8;
    for (int nb = 0; nb < ND; nb += 32) {
      __syncthreads();
#pragma unroll
      for (int j = 0; j < 8; ++j) {
        int k = kq + j;
        sB[b_idx(k, bo)]        = (__bf16)hW[(size_t)(WROWS + N0 + bo) * ND + nb + k];
        sB[2048 + b_idx(k, bo)] = (__bf16)pW[(size_t)(WROWS + N0 + bo) * ND + nb + k];
        float zv = z[(M0 + bo) * ND + nb + k];
        sA[a_idx(bo, k)]        = (__bf16)zv;
        sA[2048 + a_idx(bo, k)] = (__bf16)zv;
      }
      __syncthreads();
      mma_stage_16x32(sA, sB, wm, wn, lane, acc0, acc1);
    }
  }
  // scalar bias tail (added exactly once per output element by this kernel)
  const int ncol  = N0 + wn * 32 + (lane & 15);
  const int mbase = M0 + wm * 16 + ((lane >> 4) << 3);
  const float t0 = hb[WROWS + ncol]      + pb[WROWS + ncol];
  const float t1 = hb[WROWS + ncol + 16] + pb[WROWS + ncol + 16];
#pragma unroll
  for (int r = 0; r < 8; ++r) {
    atomicAdd(&out[(mbase + r) * OUT_F + ncol],      acc0[r] + t0);
    atomicAdd(&out[(mbase + r) * OUT_F + ncol + 16], acc1[r] + t1);
  }
}

extern "C" void kernel_launch(void* const* d_in, const int* in_sizes, int n_in,
                              void* d_out, int out_size, void* d_ws, size_t ws_size,
                              hipStream_t stream) {
  const float* x  = (const float*)d_in[0];   // (512, 512)
  const float* px = (const float*)d_in[1];   // (512, 512)
  const float* z  = (const float*)d_in[2];   // (512, 128)
  const float* hW = (const float*)d_in[3];   // (262656, 128)
  const float* hb = (const float*)d_in[4];   // (262656,)
  const float* pW = (const float*)d_in[5];   // (262656, 128)
  const float* pb = (const float*)d_in[6];   // (262656,)
  float* out = (float*)d_out;                // (512, 512)

  zero_out_kernel<<<(out_size + 255) / 256, 256, 0, stream>>>(out, out_size);
  hyper_main_kernel<<<dim3(4, 8, 8), 256, 0, stream>>>(x, px, z, hW, pW, out);
  hyper_extras_kernel<<<dim3(8, 8), 256, 0, stream>>>(x, px, z, hW, pW, hb, pb, out);
}
